// BlocksparseFixedSelfAttention_54631984005780
// MI455X (gfx1250) — matrix operand order repaired and checked by reading, not compiled, not run on hardware
//
#include <hip/hip_runtime.h>

typedef __bf16 bf16_t;
typedef __attribute__((ext_vector_type(16))) __bf16 v16bf;
typedef __attribute__((ext_vector_type(8)))  __bf16 v8bf;
typedef __attribute__((ext_vector_type(4)))  __bf16 v4bf;
typedef __attribute__((ext_vector_type(8)))  float  v8f;

#define T_SEQ 2048
#define EMB   512
#define NB    4
#define MTOT  (NB * T_SEQ)   // 8192
#define KBLK  64

// ---------------------------------------------------------------- helpers

__device__ __forceinline__ v8f v8f_zero() {
  v8f z;
#pragma unroll
  for (int i = 0; i < 8; ++i) z[i] = 0.0f;
  return z;
}

__device__ __forceinline__ v8f wmma_bf16(v16bf a, v16bf b, v8f c) {
  return __builtin_amdgcn_wmma_f32_16x16x32_bf16(
      /*neg_a=*/false, a, /*neg_b=*/false, b,
      /*c_mod=*/(short)0, c, /*reuse_a=*/false, /*reuse_b=*/false);
}

// A operand: rows [0..15] of X (row-major, ld), K tile [k0..k0+31].
// Layout used here (contiguous k, same as B): lane (h=lane>>4, m=lane&15), element e holds
//   K = k0 + 16h + e
__device__ __forceinline__ v16bf load_a(const bf16_t* X, int ld, int k0, int lane) {
  int h = (lane >> 4) & 1, m = lane & 15;
  const bf16_t* p = X + (size_t)m * ld + k0 + 16 * h;
  v8bf lo = *(const v8bf*)(p);
  v8bf hi = *(const v8bf*)(p + 8);
  v16bf a;
#pragma unroll
  for (int e = 0; e < 8; ++e) { a[e] = lo[e]; a[8 + e] = hi[e]; }
  return a;
}

// B operand = transposed access of row-major W: B[k][n] = W[(n0+n)*rowmul][k0+k]
// lane (h,n) element e holds K = k0 + 16h + e  -> one contiguous 32B load.
__device__ __forceinline__ v16bf load_b_T(const bf16_t* W, int ld, int n0, int k0,
                                          int rowmul, int lane) {
  int h = (lane >> 4) & 1, n = lane & 15;
  return *(const v16bf*)(W + (size_t)((n0 + n) * rowmul) * ld + k0 + 16 * h);
}

// CDNA5 async copy: 16B global -> LDS, tracked with ASYNCcnt
__device__ __forceinline__ void async_g2l_b128(uint32_t lds_off, const bf16_t* g) {
  asm volatile("global_load_async_to_lds_b128 %0, %1, off"
               :: "v"(lds_off), "v"(g) : "memory");
}
__device__ __forceinline__ void wait_async0() {
  asm volatile("s_wait_asynccnt 0x0" ::: "memory");
}

// ---------------------------------------------------------------- small kernels

__global__ void cvt_f32_bf16(const float* __restrict__ in, bf16_t* __restrict__ out,
                             int n) {
  int i = (blockIdx.x * blockDim.x + threadIdx.x) * 4;
  if (i < n) {
    float4 f = *(const float4*)(in + i);
    v4bf o;
    o[0] = (bf16_t)f.x; o[1] = (bf16_t)f.y; o[2] = (bf16_t)f.z; o[3] = (bf16_t)f.w;
    *(v4bf*)(out + i) = o;
  }
}

// VtT[b][e][t] = Vt[b][t][e], 64x64 LDS-tiled transpose
__global__ __launch_bounds__(256) void transpose_bf16(
    const bf16_t* __restrict__ in, bf16_t* __restrict__ out) {
  __shared__ bf16_t tile[64 * 65];
  int t0 = blockIdx.x * 64, e0 = blockIdx.y * 64, b = blockIdx.z;
  const bf16_t* ib = in + (size_t)b * T_SEQ * EMB;
  bf16_t* ob = out + (size_t)b * EMB * T_SEQ;
#pragma unroll
  for (int i = 0; i < 16; ++i) {
    int idx = threadIdx.x + i * 256;
    int r = idx >> 6, c = idx & 63;
    tile[r * 65 + c] = ib[(size_t)(t0 + r) * EMB + e0 + c];
  }
  __syncthreads();
#pragma unroll
  for (int i = 0; i < 16; ++i) {
    int idx = threadIdx.x + i * 256;
    int r = idx >> 6, c = idx & 63;  // r: e-local, c: t-local
    ob[(size_t)(e0 + r) * T_SEQ + t0 + c] = tile[c * 65 + r];
  }
}

// VtST[b][e][c] = Vt[b][64c][e]  (block-start rows of Vt, transposed; 512x32/batch)
__global__ void gather_vtst(const bf16_t* __restrict__ Vt, bf16_t* __restrict__ VtST) {
  int idx = blockIdx.x * blockDim.x + threadIdx.x;  // NB*512*32
  if (idx >= NB * EMB * 32) return;
  int c = idx & 31, e = (idx >> 5) & (EMB - 1), b = idx >> 14;
  VtST[idx] = Vt[(size_t)b * T_SEQ * EMB + (size_t)(64 * c) * EMB + e];
}

// ---------------------------------------------------------------- main GEMM
// C[M,N] = A[M,K] @ W[N,K]^T + bias.  Block: 8 waves, wave tile 32x64, block 256x64.
// W panel (64 x 256-k chunk, 32KB) staged into LDS with async-to-LDS copies.
template <int K, bool OUT_F32>
__global__ __launch_bounds__(256) void gemm_ws(
    const bf16_t* __restrict__ A, const bf16_t* __restrict__ W,
    const float* __restrict__ bias, void* __restrict__ Cout, int ldc) {
  __shared__ __align__(16) bf16_t Wlds[64 * 256];
  int lane = threadIdx.x & 31, wv = threadIdx.x >> 5;
  int row0 = blockIdx.x * 256 + wv * 32;
  int n0 = blockIdx.y * 64;
  const bf16_t* Ar = A + (size_t)row0 * K;

  v8f acc[2][4];
#pragma unroll
  for (int rt = 0; rt < 2; ++rt)
#pragma unroll
    for (int t = 0; t < 4; ++t) acc[rt][t] = v8f_zero();

  for (int kc = 0; kc < K; kc += 256) {
    // ---- async stage W[n0..n0+64) x [kc..kc+256) into LDS (2048 x b128)
    {
      int t = threadIdx.x;
#pragma unroll
      for (int i = 0; i < 8; ++i) {
        int unit = t * 8 + i;        // 32 b128-units per 256-elem row
        int r = unit >> 5;
        int c = (unit & 31) * 8;
        async_g2l_b128((uint32_t)(uintptr_t)&Wlds[r * 256 + c],
                       W + (size_t)(n0 + r) * K + kc + c);
      }
      wait_async0();
    }
    __syncthreads();
    // ---- compute on the chunk: B operands now come from LDS (8-wave reuse)
    for (int k0 = 0; k0 < 256; k0 += 32) {
      v16bf a0 = load_a(Ar, K, kc + k0, lane);
      v16bf a1 = load_a(Ar + (size_t)16 * K, K, kc + k0, lane);
#pragma unroll
      for (int t = 0; t < 4; ++t) {
        v16bf b = load_b_T(Wlds, 256, 16 * t, k0, 1, lane);
        acc[0][t] = wmma_bf16(a0, b, acc[0][t]);
        acc[1][t] = wmma_bf16(a1, b, acc[1][t]);
      }
    }
    __syncthreads();
  }

  int h = (lane >> 4) & 1, n = lane & 15;
#pragma unroll
  for (int rt = 0; rt < 2; ++rt)
#pragma unroll
    for (int t = 0; t < 4; ++t) {
      int col = n0 + 16 * t + n;
      float bb = bias[col];
#pragma unroll
      for (int v = 0; v < 8; ++v) {
        int row = row0 + rt * 16 + 8 * h + v;
        float val = acc[rt][t][v] + bb;
        if constexpr (OUT_F32)
          ((float*)Cout)[(size_t)row * ldc + col] = val;
        else
          ((bf16_t*)Cout)[(size_t)row * ldc + col] = (bf16_t)val;
      }
    }
}

// ---------------------------------------------------------------- heads

// head1: per (block, batch): S = tril(Kt_blk @ Qt_blk^T) (64x64), h1 = S @ Vt_blk
__global__ __launch_bounds__(128) void head1_kernel(
    const bf16_t* __restrict__ Kt, const bf16_t* __restrict__ Qt,
    const bf16_t* __restrict__ VtT, bf16_t* __restrict__ comb) {
  __shared__ __align__(16) bf16_t Slds[64 * 80];
  int s = blockIdx.x * KBLK;
  int b = blockIdx.y;
  int lane = threadIdx.x & 31, wv = threadIdx.x >> 5;
  const bf16_t* Kb = Kt + ((size_t)b * T_SEQ + s) * EMB;
  const bf16_t* Qb = Qt + ((size_t)b * T_SEQ + s) * EMB;
  const bf16_t* VtTb = VtT + (size_t)b * EMB * T_SEQ;

  v8f acc[4];
#pragma unroll
  for (int t = 0; t < 4; ++t) acc[t] = v8f_zero();
  const bf16_t* Ar = Kb + (size_t)(16 * wv) * EMB;
  for (int k0 = 0; k0 < EMB; k0 += 32) {
    v16bf a = load_a(Ar, EMB, k0, lane);
#pragma unroll
    for (int ct = 0; ct < 4; ++ct) {
      v16bf bb = load_b_T(Qb, EMB, 16 * ct, k0, 1, lane);
      acc[ct] = wmma_bf16(a, bb, acc[ct]);
    }
  }

  int h = (lane >> 4) & 1, n = lane & 15;
#pragma unroll
  for (int ct = 0; ct < 4; ++ct)
#pragma unroll
    for (int v = 0; v < 8; ++v) {
      int r = 16 * wv + 8 * h + v;
      int c = 16 * ct + n;
      float val = (c <= r) ? acc[ct][v] : 0.0f;  // lower-triangular mask
      Slds[r * 80 + c] = (bf16_t)val;
    }
  __syncthreads();

  // h1 strip = S(16x64) @ V(64x512); B from VtT -> contiguous 32B loads
  for (int n0 = 0; n0 < EMB; n0 += 64) {
    v8f o[4];
#pragma unroll
    for (int t = 0; t < 4; ++t) o[t] = v8f_zero();
#pragma unroll
    for (int k0 = 0; k0 < KBLK; k0 += 32) {
      v16bf a = load_a(Slds + 16 * wv * 80, 80, k0, lane);
#pragma unroll
      for (int ct = 0; ct < 4; ++ct) {
        v16bf bb = load_b_T(VtTb, T_SEQ, n0 + 16 * ct, s + k0, 1, lane);
        o[ct] = wmma_bf16(a, bb, o[ct]);
      }
    }
#pragma unroll
    for (int ct = 0; ct < 4; ++ct)
#pragma unroll
      for (int v = 0; v < 8; ++v) {
        int r = 16 * wv + 8 * h + v;
        int c = n0 + 16 * ct + n;
        comb[((size_t)b * T_SEQ + s + r) * (2 * EMB) + c] = (bf16_t)o[ct][v];
      }
  }
}

// head2: S2 = mask(Kt @ Qt_starts^T) (2048x32), h2 = S2 @ Vt_starts (32x512)
__global__ __launch_bounds__(128) void head2_kernel(
    const bf16_t* __restrict__ Kt, const bf16_t* __restrict__ Qt,
    const bf16_t* __restrict__ VtST, bf16_t* __restrict__ comb) {
  __shared__ __align__(16) bf16_t Slds[64 * 48];
  int grp = blockIdx.x;
  int b = blockIdx.y;
  int lane = threadIdx.x & 31, wv = threadIdx.x >> 5;
  const bf16_t* Ktb = Kt + (size_t)b * T_SEQ * EMB;
  const bf16_t* Qtb = Qt + (size_t)b * T_SEQ * EMB;
  const bf16_t* VtSTb = VtST + (size_t)b * EMB * 32;

  v8f acc[2];
  acc[0] = v8f_zero();
  acc[1] = v8f_zero();
  const bf16_t* Ar = Ktb + (size_t)(grp * 64 + wv * 16) * EMB;
  for (int k0 = 0; k0 < EMB; k0 += 32) {
    v16bf a = load_a(Ar, EMB, k0, lane);
#pragma unroll
    for (int ct = 0; ct < 2; ++ct) {
      v16bf bb = load_b_T(Qtb, EMB, 16 * ct, k0, KBLK, lane);  // B[k][c]=Qt[64c][k]
      acc[ct] = wmma_bf16(a, bb, acc[ct]);
    }
  }

  int h = (lane >> 4) & 1, n = lane & 15;
#pragma unroll
  for (int ct = 0; ct < 2; ++ct)
#pragma unroll
    for (int v = 0; v < 8; ++v) {
      int rl = 16 * wv + 8 * h + v;
      int c = 16 * ct + n;
      int r = grp * 64 + rl;
      float val = (c * KBLK <= r) ? acc[ct][v] : 0.0f;
      Slds[rl * 48 + c] = (bf16_t)val;
    }
  __syncthreads();

  // h2 strip = S2(16x32) @ Vt_starts(32x512); B from VtST -> contiguous loads
  v16bf a2 = load_a(Slds + 16 * wv * 48, 48, 0, lane);
  for (int n0 = 0; n0 < EMB; n0 += 64) {
    v8f o[4];
#pragma unroll
    for (int t = 0; t < 4; ++t) o[t] = v8f_zero();
#pragma unroll
    for (int ct = 0; ct < 4; ++ct) {
      v16bf bb = load_b_T(VtSTb, 32, n0 + 16 * ct, 0, 1, lane);
      o[ct] = wmma_bf16(a2, bb, o[ct]);
    }
#pragma unroll
    for (int ct = 0; ct < 4; ++ct)
#pragma unroll
      for (int v = 0; v < 8; ++v) {
        int r = grp * 64 + 16 * wv + 8 * h + v;
        int c = n0 + 16 * ct + n;
        comb[((size_t)b * T_SEQ + r) * (2 * EMB) + EMB + c] = (bf16_t)o[ct][v];
      }
  }
}

// ---------------------------------------------------------------- launch

extern "C" void kernel_launch(void* const* d_in, const int* in_sizes, int n_in,
                              void* d_out, int out_size, void* d_ws, size_t ws_size,
                              hipStream_t stream) {
  (void)in_sizes; (void)n_in; (void)out_size; (void)ws_size;
  const float* x  = (const float*)d_in[0];
  const float* Wk = (const float*)d_in[1];
  const float* bk = (const float*)d_in[2];
  const float* Wq = (const float*)d_in[3];
  const float* bq = (const float*)d_in[4];
  const float* Wv = (const float*)d_in[5];
  const float* bv = (const float*)d_in[6];
  const float* Wu = (const float*)d_in[7];
  const float* bu = (const float*)d_in[8];
  float* out = (float*)d_out;

  char* ws = (char*)d_ws;
  size_t off = 0;
  auto take = [&](size_t elems) -> bf16_t* {
    bf16_t* p = (bf16_t*)(ws + off);
    off += ((elems * sizeof(bf16_t)) + 255) & ~(size_t)255;
    return p;
  };
  bf16_t* xb   = take((size_t)MTOT * EMB);
  bf16_t* Wkb  = take((size_t)EMB * EMB);
  bf16_t* Wqb  = take((size_t)EMB * EMB);
  bf16_t* Wvb  = take((size_t)EMB * EMB);
  bf16_t* Wub  = take((size_t)EMB * 2 * EMB);
  bf16_t* Ktb  = take((size_t)MTOT * EMB);
  bf16_t* Qtb  = take((size_t)MTOT * EMB);
  bf16_t* Vtb  = take((size_t)MTOT * EMB);
  bf16_t* VtT  = take((size_t)MTOT * EMB);       // [NB][EMB][T_SEQ]
  bf16_t* VtST = take((size_t)NB * EMB * 32);    // [NB][EMB][32]
  bf16_t* comb = take((size_t)MTOT * 2 * EMB);

  auto cvt = [&](const float* src, bf16_t* dst, int n) {
    cvt_f32_bf16<<<(n / 4 + 255) / 256, 256, 0, stream>>>(src, dst, n);
  };
  cvt(x, xb, MTOT * EMB);
  cvt(Wk, Wkb, EMB * EMB);
  cvt(Wq, Wqb, EMB * EMB);
  cvt(Wv, Wvb, EMB * EMB);
  cvt(Wu, Wub, EMB * 2 * EMB);

  dim3 ggrid(MTOT / 256, EMB / 64);
  gemm_ws<EMB, false><<<ggrid, 256, 0, stream>>>(xb, Wkb, bk, Ktb, EMB);
  gemm_ws<EMB, false><<<ggrid, 256, 0, stream>>>(xb, Wqb, bq, Qtb, EMB);
  gemm_ws<EMB, false><<<ggrid, 256, 0, stream>>>(xb, Wvb, bv, Vtb, EMB);

  transpose_bf16<<<dim3(T_SEQ / 64, EMB / 64, NB), 256, 0, stream>>>(Vtb, VtT);
  gather_vtst<<<(NB * EMB * 32 + 255) / 256, 256, 0, stream>>>(Vtb, VtST);

  dim3 hgrid(T_SEQ / KBLK, NB);
  head1_kernel<<<hgrid, 128, 0, stream>>>(Ktb, Qtb, VtT, comb);
  head2_kernel<<<hgrid, 128, 0, stream>>>(Ktb, Qtb, VtST, comb);

  gemm_ws<2 * EMB, true><<<ggrid, 256, 0, stream>>>(comb, Wub, bu, out, EMB);
}
